// AttentionAggregator_57019985822575
// MI455X (gfx1250) — compile-verified, matrix operand order checked by reading, not run on hardware
//
#include <hip/hip_runtime.h>
#include <hip/hip_bf16.h>
#include <math.h>

// ---------------------------------------------------------------------------
// AttentionAggregator fused kernel for MI455X (gfx1250, wave32, WMMA).
// B=16384, G=50, D=256.  GEMM in bf16 WMMA (16x16x32, f32 accum), everything
// else fused in LDS.  One WG = 256 threads = 8 wave32, handles NB=8 batches.
// W1 and the x tile live in LDS pre-swizzled in WMMA fragment order:
// per 16x32 tile (1KB contiguous), lane's lo chunk at base+lane*16B and hi
// chunk at base+512+lane*16B -> each ds_load_b128 touches 128 consecutive
// dwords (2 accesses/bank, conflict-free, no padding).
// Each wave computes a 2 M-tile x 4 N-tile block: 12 ds_load_b128 per 8 WMMA,
// with all fragments loaded before the WMMA burst so waits stagger.
// ---------------------------------------------------------------------------

#define BATCH   16384
#define GRP     50
#define DIM     256
#define NB      8          // batches per workgroup
#define NWG     (BATCH / NB)

// LDS layout (bytes).
#define OFF_SW      0                              // W1 swizzled: 128 tiles*1KB
#define OFF_SA      (OFF_SW + 131072)              // x swizzled: 32 tiles*1KB
#define OFF_SH      (OFF_SA + 32768)               // h fp32: 64 x 260 (padded)
#define SH_STRIDE   260
#define OFF_SWA     (OFF_SH + 64 * SH_STRIDE * 4)  // 230400
#define OFF_SB1     (OFF_SWA + 1024)               // 231424
#define OFF_SRED    (OFF_SB1 + 1024)               // 232448: 512 floats
#define SMEM_BYTES  (OFF_SRED + 2048)              // 234496

typedef __attribute__((ext_vector_type(16))) __bf16 v16bf;
typedef __attribute__((ext_vector_type(8)))  __bf16 v8bf;
typedef __attribute__((ext_vector_type(8)))  float  v8f;

static __device__ __forceinline__ unsigned int f2bf(float f) {
    unsigned u = __float_as_uint(f);
    return (u + 0x7FFFu + ((u >> 16) & 1u)) >> 16;   // RNE, inputs finite
}
static __device__ __forceinline__ unsigned int pack2(float lo, float hi) {
    return f2bf(lo) | (f2bf(hi) << 16);
}

// WMMA-fragment-order swizzle: tile = 16 rows x 32 K of bf16, 512 elems.
// within-tile u16 offset = c*256 + lane*8 + e, where
//   lane = half*16 + row15, K = kk*32 + c*16 + half*8 + e.
static __device__ __forceinline__ int frag_off(int tile, int row15, int K) {
    int c    = (K >> 4) & 1;
    int half = (K >> 3) & 1;
    int e    = K & 7;
    return tile * 512 + c * 256 + (half * 16 + row15) * 8 + e;
}

static __device__ __forceinline__ v16bf load_frag(const unsigned short* base) {
    v8bf lo = *(const v8bf*)(base);
    v8bf hi = *(const v8bf*)(base + 256);
    return __builtin_shufflevector(lo, hi,
        0,1,2,3,4,5,6,7,8,9,10,11,12,13,14,15);
}

// ------------- prologue: W1 fp32 -> bf16, swizzled to fragment order --------
__global__ __launch_bounds__(256) void w1_swizzle_kernel(
    const float* __restrict__ W1, unsigned short* __restrict__ out) {
    int ci = blockIdx.x * 256 + threadIdx.x;        // one 8-elem chunk each
    if (ci >= DIM * DIM / 8) return;                // 8192 chunks
    int n = ci >> 5;                                // output feature (B column)
    int K = (ci & 31) * 8;                          // input feature
    const float4* src = (const float4*)(W1 + n * DIM + K);
    float4 p0 = src[0], p1 = src[1];
    int tile = (n >> 4) * 8 + (K >> 5);             // nt*8 + kk
    int off  = frag_off(tile, n & 15, K & 31);      // e==0 here
    uint4 v;
    v.x = pack2(p0.x, p0.y);  v.y = pack2(p0.z, p0.w);
    v.z = pack2(p1.x, p1.y);  v.w = pack2(p1.z, p1.w);
    *(uint4*)(out + off) = v;
}

// ------------------------------ main fused kernel ---------------------------
__global__ __launch_bounds__(256) void attn_agg_kernel(
    const float* __restrict__ x,            // [B,G,D]
    const float* __restrict__ mask,         // [B,G]
    const unsigned short* __restrict__ W1s, // swizzled bf16 [128 tiles * 512]
    const float* __restrict__ b1,           // [D]
    const float* __restrict__ wa,           // [D]
    const float* __restrict__ ba,           // [1]
    const float* __restrict__ Wc,           // [2]
    const float* __restrict__ bc,           // [2]
    const float* __restrict__ bias,         // [1]
    float* __restrict__ out)                // ret | weight | pred
{
    extern __shared__ char smem[];
    unsigned short* sW  = (unsigned short*)(smem + OFF_SW);
    unsigned short* sA  = (unsigned short*)(smem + OFF_SA);
    float*          sH  = (float*)(smem + OFF_SH);
    float*          sWa = (float*)(smem + OFF_SWA);
    float*          sB1 = (float*)(smem + OFF_SB1);
    float*          sRed= (float*)(smem + OFF_SRED);

    const int tid  = threadIdx.x;
    const int lane = tid & 31;
    const int wave = tid >> 5;                 // 0..7
    const int mtBase = (wave >> 2) * 2;        // M-tiles mtBase, mtBase+1
    const int ntBase = (wave & 3) * 4;         // N-tiles ntBase..ntBase+3
    const int half   = lane >> 4;
    const int l15    = lane & 15;

    // ---- stage swizzled W1 into LDS once per WG: linear 128KB memcpy ----
    {
        const uint4* src = (const uint4*)W1s;
        uint4* dst = (uint4*)sW;
        for (int i = tid; i < 8192; i += 256) dst[i] = src[i];
        sWa[tid] = wa[tid];
        sB1[tid] = b1[tid];
    }

    const float ba0 = ba[0];
    const float wc0 = Wc[0], wc1 = Wc[1];
    const float bc0 = bc[0], bc1 = bc[1] + bias[0];

    for (int ib = 0; ib < NB; ++ib) {
        const int b = blockIdx.x * NB + ib;
        __syncthreads();   // protect LDS reuse across batches (and W1 stage)

        // ---- stage x[b] -> bf16, swizzled; coalesced float4 reads,
        //      one ds_store_b128 per 8-element chunk ----
        {
            const float4* xb4 = (const float4*)(x + (size_t)b * (GRP * DIM));
            for (int ci = tid; ci < GRP * DIM / 8; ci += 256) {  // 1600 chunks
                int m = ci >> 5;                 // row (g), always < 50
                int K = (ci & 31) * 8;
                float4 p0 = xb4[ci * 2], p1 = xb4[ci * 2 + 1];
                int tile = (m >> 4) * 8 + (K >> 5);      // mt*8 + kk
                int off  = frag_off(tile, m & 15, K & 31);
                uint4 v;
                v.x = pack2(p0.x, p0.y);  v.y = pack2(p0.z, p0.w);
                v.z = pack2(p1.x, p1.y);  v.w = pack2(p1.z, p1.w);
                *(uint4*)(sA + off) = v;
            }
        }

        // ---- prefetch next batch's x into cache while we compute ----
        if (ib + 1 < NB) {
            const char* xn = (const char*)(x + (size_t)(b + 1) * (GRP * DIM));
            for (int l = tid; l < GRP * DIM * 4 / 128; l += 256)   // 400 lines
                __builtin_prefetch(xn + l * 128, 0, 3);
        }
        __syncthreads();

        // ---- GEMM: h(64x256) = x(64x256) * W1^T, bf16 WMMA f32-acc ----
        v8f acc0[4], acc1[4];
        #pragma unroll
        for (int j = 0; j < 4; ++j) {
            acc0[j] = (v8f){0.f,0.f,0.f,0.f,0.f,0.f,0.f,0.f};
            acc1[j] = (v8f){0.f,0.f,0.f,0.f,0.f,0.f,0.f,0.f};
        }

        #pragma unroll
        for (int kk = 0; kk < 8; ++kk) {       // K = 8 * 32
            // Load ALL fragments for this K-step first, then WMMA burst,
            // so dscnt waits stagger and overlap WMMA execution.
            v16bf a0 = load_frag(sA + ((mtBase    ) * 8 + kk) * 512 + lane * 8);
            v16bf a1 = load_frag(sA + ((mtBase + 1) * 8 + kk) * 512 + lane * 8);
            v16bf bf[4];
            #pragma unroll
            for (int j = 0; j < 4; ++j)
                bf[j] = load_frag(sW + ((ntBase + j) * 8 + kk) * 512 + lane * 8);

            #pragma unroll
            for (int j = 0; j < 4; ++j) {
                acc0[j] = __builtin_amdgcn_wmma_f32_16x16x32_bf16(
                    false, a0, false, bf[j], (short)0, acc0[j], false, false);
                acc1[j] = __builtin_amdgcn_wmma_f32_16x16x32_bf16(
                    false, a1, false, bf[j], (short)0, acc1[j], false, false);
            }
        }

        // ---- epilogue: h = tanh(relu(acc + b1)) -> LDS ----
        #pragma unroll
        for (int j = 0; j < 4; ++j) {
            const int n = (ntBase + j) * 16 + l15;
            const float bn = sB1[n];
            #pragma unroll
            for (int e = 0; e < 8; ++e) {
                const int m0 = mtBase * 16 + half * 8 + e;  // C/D VGPR layout
                float v0 = acc0[j][e] + bn;
                v0 = v0 > 0.f ? v0 : 0.f;
                sH[m0 * SH_STRIDE + n] = tanhf(v0);
                float v1 = acc1[j][e] + bn;
                v1 = v1 > 0.f ? v1 : 0.f;
                sH[(m0 + 16) * SH_STRIDE + n] = tanhf(v1);
            }
        }
        __syncthreads();

        // ---- attention logits, parallel: 4 segments x 64 rows ----
        {
            int seg = tid >> 6, g = tid & 63;
            float p = 0.f;
            if (g < GRP) {
                const float* hr = sH + g * SH_STRIDE + seg * 64;
                const float* wr = sWa + seg * 64;
                for (int d = 0; d < 64; ++d) p = fmaf(hr[d], wr[d], p);
            }
            sRed[tid] = p;
        }
        __syncthreads();
        if (tid < GRP) {
            float s = ba0 + sRed[tid] + sRed[64 + tid]
                          + sRed[128 + tid] + sRed[192 + tid];
            float a  = tanhf(s);
            float mk = mask[(size_t)b * GRP + tid];
            sRed[256 + tid] = a + mk;                                  // logit
            sRed[320 + tid] = (mk > -INFINITY && mk < INFINITY) ? 1.f : 0.f;
        }
        __syncthreads();

        // ---- softmax / variance / classifier / argmax (tiny, serial) ----
        if (tid == 0) {
            float mx = -INFINITY;
            for (int g = 0; g < GRP; ++g) mx = fmaxf(mx, sRed[256 + g]);
            float sum = 0.f;
            for (int g = 0; g < GRP; ++g) {
                float e = __expf(sRed[256 + g] - mx);
                sRed[384 + g] = e;  sum += e;
            }
            float inv = 1.f / sum, n = 0.f, msum = 0.f;
            for (int g = 0; g < GRP; ++g) {
                float w = sRed[384 + g] * inv;  sRed[384 + g] = w;
                float p = sRed[320 + g];
                n += p;  msum += w * p;
            }
            float mean = msum / n, var = 0.f;
            for (int g = 0; g < GRP; ++g) {
                float p = sRed[320 + g];
                float d = sRed[384 + g] - mean;
                var += p * d * d;
            }
            var /= (n - 1.f);
            float s0 = wc0 * var + bc0;
            float s1 = wc1 * var + bc1;
            int cls = (s1 > s0) ? 1 : 0;                 // argmax, first-max tie
            float bw = -INFINITY; int ti = 0;
            for (int g = 0; g < GRP; ++g)
                if (sRed[384 + g] > bw) { bw = sRed[384 + g]; ti = g; }
            sRed[448] = (float)cls;
            ((int*)sRed)[449] = ti;
        }
        __syncthreads();

        // ---- outputs ----
        {
            const int cls = (sRed[448] != 0.f);
            const int ti  = ((const int*)sRed)[449];
            const int d   = tid;
            float ws = 0.f;
            for (int g = 0; g < GRP; ++g)
                ws = fmaf(sRed[384 + g], sH[g * SH_STRIDE + d], ws);
            float r = cls ? sH[ti * SH_STRIDE + d] : ws;
            out[(size_t)b * DIM + d] = r;                                 // ret
            if (tid < GRP)
                out[(size_t)BATCH * DIM + (size_t)b * GRP + tid] = sRed[384 + tid];
            if (tid == 0)
                out[(size_t)BATCH * DIM + (size_t)BATCH * GRP + b] = sRed[448];
        }
    }
}

// ---------------------------------------------------------------------------
extern "C" void kernel_launch(void* const* d_in, const int* in_sizes, int n_in,
                              void* d_out, int out_size, void* d_ws, size_t ws_size,
                              hipStream_t stream) {
    const float* x    = (const float*)d_in[0];
    const float* mask = (const float*)d_in[1];
    const float* W1   = (const float*)d_in[2];
    const float* b1   = (const float*)d_in[3];
    const float* wa   = (const float*)d_in[4];
    const float* ba   = (const float*)d_in[5];
    const float* Wc   = (const float*)d_in[6];
    const float* bc   = (const float*)d_in[7];
    const float* bias = (const float*)d_in[8];
    float* out = (float*)d_out;

    unsigned short* w1s = (unsigned short*)d_ws;    // 128 KB swizzled W1

    w1_swizzle_kernel<<<(DIM * DIM / 8 + 255) / 256, 256, 0, stream>>>(W1, w1s);

    (void)hipFuncSetAttribute((const void*)attn_agg_kernel,
                              hipFuncAttributeMaxDynamicSharedMemorySize,
                              SMEM_BYTES);
    attn_agg_kernel<<<NWG, 256, SMEM_BYTES, stream>>>(
        x, mask, w1s, b1, wa, ba, Wc, bc, bias, out);
}
